// BertSelfAttention_20143396618552
// MI455X (gfx1250) — matrix operand order repaired and checked by reading, not compiled, not run on hardware
//
#include <hip/hip_runtime.h>

// ---------------------------------------------------------------------------
// BERT self-attention for MI455X (gfx1250, wave32, WMMA + async-to-LDS DMA).
//   B=8, S=1024, H=1024, NH=16, HD=64.  Reference I/O fp32.
// Pipeline:
//   1) one-shot f32->bf16 convert of HS, and f32->bf16 transpose of W
//   2) QKV GEMM: bf16 tiles staged by GLOBAL_LOAD_ASYNC_TO_LDS_B128
//      (double-buffered), V_WMMA_F32_16X16X32_BF16, per-head scatter (V^T)
//   3) flash attention: K / V^T tiles async-staged into LDS shared by 8
//      waves, online softmax, WMMA for QK^T and PV.
// Workspace (bf16): Q,K,V^T (48 MB) + HS_bf16 (16 MB) + W^T_bf16 (6 MB).
// Everything L2-resident (192 MB L2).
// ---------------------------------------------------------------------------

#define SEQ   1024
#define HID   1024
#define N3    3072
#define HD    64
#define NHEAD 16

typedef __bf16 bf16;
typedef __attribute__((ext_vector_type(16))) __bf16 v16bf;
typedef __attribute__((ext_vector_type(8)))  __bf16 v8bf;
typedef __attribute__((ext_vector_type(8)))  float  v8f;

#define WMMA_BF16(a, b, c) \
  __builtin_amdgcn_wmma_f32_16x16x32_bf16(false, (a), false, (b), (short)0, (c), false, false)

static __device__ __forceinline__ v16bf pack16(v8bf lo, v8bf hi) {
  v16bf r;
#pragma unroll
  for (int i = 0; i < 8; ++i) { r[i] = lo[i]; r[i + 8] = hi[i]; }
  return r;
}

// A-fragment (16x32 bf16): lane = row (lane&15); K halves c0..c0+7 / c0+16..23.
static __device__ __forceinline__ v16bf load_a_frag(const bf16* base, int stride) {
  const int lane = threadIdx.x & 31;
  const bf16* p = base + (lane & 15) * stride + ((lane >> 4) * 8);
  return pack16(*(const v8bf*)p, *(const v8bf*)(p + 16));
}
// B-fragment (32x16 bf16) from N-major storage: lane = column; K halves c0..c0+7 / c0+16..23.
static __device__ __forceinline__ v16bf load_b_fragT(const bf16* base, int stride) {
  const int lane = threadIdx.x & 31;
  const bf16* p = base + (lane & 15) * stride + ((lane >> 4) * 8);
  return pack16(*(const v8bf*)p, *(const v8bf*)(p + 16));
}

// Async DMA: 16 bytes/lane global -> LDS, tracked by ASYNCcnt (ISA 08).
// LDS byte address = low 32 bits of the generic shared pointer.
static __device__ __forceinline__ void async_cp16(const bf16* lds_dst, const bf16* gsrc) {
  const unsigned lds_addr = (unsigned)(size_t)lds_dst;
  asm volatile("global_load_async_to_lds_b128 %0, %1, off"
               :: "v"(lds_addr), "v"(gsrc) : "memory");
}
static __device__ __forceinline__ void wait_async0() {
  asm volatile("s_wait_asynccnt 0x0" ::: "memory");
}

// ---------------------------------------------------------------------------
// Prep 1: HS f32 -> bf16 (same layout).  Grid 4096 x 256.
// ---------------------------------------------------------------------------
__global__ __launch_bounds__(256)
void cvt_hs_kernel(const float* __restrict__ hs, bf16* __restrict__ Abf) {
  const size_t i = ((size_t)blockIdx.x * 256 + threadIdx.x) * 8;
  const float4 a = *(const float4*)(hs + i);
  const float4 b = *(const float4*)(hs + i + 4);
  v8bf o;
  o[0] = (bf16)a.x; o[1] = (bf16)a.y; o[2] = (bf16)a.z; o[3] = (bf16)a.w;
  o[4] = (bf16)b.x; o[5] = (bf16)b.y; o[6] = (bf16)b.z; o[7] = (bf16)b.w;
  *(v8bf*)(Abf + i) = o;
}

// ---------------------------------------------------------------------------
// Prep 2: W (k-major f32) -> W^T (n-major bf16) via LDS tile.  Grid (48,16).
// ---------------------------------------------------------------------------
__global__ __launch_bounds__(256)
void wt_kernel(const float* __restrict__ w, bf16* __restrict__ Wt) {
  constexpr int TS = 72;
  __shared__ bf16 T[64 * TS];
  const int t  = threadIdx.x;
  const int n0 = blockIdx.x * 64;
  const int k0 = blockIdx.y * 64;
#pragma unroll
  for (int i = 0; i < 16; ++i) {
    const int idx = t + i * 256;            // 0..4095
    const int kk = idx >> 6, nn = idx & 63; // coalesced along n
    T[nn * TS + kk] = (bf16)w[(size_t)(k0 + kk) * N3 + n0 + nn];
  }
  __syncthreads();
#pragma unroll
  for (int i = 0; i < 2; ++i) {
    const int c = t + i * 256;              // 0..511 16B chunks
    const int nn = c >> 3, kc = (c & 7) * 8;
    *(v8bf*)(Wt + (size_t)(n0 + nn) * HID + k0 + kc) = *(const v8bf*)&T[nn * TS + kc];
  }
}

// ---------------------------------------------------------------------------
// Kernel 1: QKV = HS @ W + b  ->  bf16 Q/K ([bh][s][d]) and V^T ([bh][d][s]).
// Grid (48,128), 128 threads = 4 waves, double-buffered async LDS staging.
// ---------------------------------------------------------------------------
__global__ __launch_bounds__(128)
void qkv_gemm_kernel(const bf16* __restrict__ Abf, const bf16* __restrict__ Wt,
                     const float* __restrict__ bias,
                     bf16* __restrict__ Qbf, bf16* __restrict__ Kbf,
                     bf16* __restrict__ Vt) {
  constexpr int LSTR = 40;                 // padded: conflict-free b128 frag reads
  __shared__ bf16 At[2][64 * LSTR];
  __shared__ bf16 Bt[2][64 * LSTR];
  const int t    = threadIdx.x;
  const int lane = t & 31;
  const int wave = t >> 5;
  const int m0 = blockIdx.y * 64;
  const int n0 = blockIdx.x * 64;
  const int wm = (wave >> 1) * 32;
  const int wn = (wave & 1) * 32;

  // Stage one 64x32 A tile + 64x32 B tile (16B chunks, 4 asyncs per thread).
  auto stage = [&](int buf, int k0) {
#pragma unroll
    for (int i = 0; i < 2; ++i) {
      const int c = t + i * 128;            // 0..255
      const int row = c >> 2, cq = (c & 3) * 8;
      async_cp16(&At[buf][row * LSTR + cq], Abf + (size_t)(m0 + row) * HID + k0 + cq);
    }
#pragma unroll
    for (int i = 0; i < 2; ++i) {
      const int c = t + i * 128;
      const int row = c >> 2, cq = (c & 3) * 8;
      async_cp16(&Bt[buf][row * LSTR + cq], Wt + (size_t)(n0 + row) * HID + k0 + cq);
    }
  };

  v8f acc[2][2];
#pragma unroll
  for (int i = 0; i < 2; ++i)
#pragma unroll
    for (int j = 0; j < 2; ++j)
#pragma unroll
      for (int e = 0; e < 8; ++e) acc[i][j][e] = 0.f;

  stage(0, 0);
  for (int kt = 0; kt < HID / 32; ++kt) {
    const int buf = kt & 1;
    wait_async0();                          // previous group only
    __syncthreads();                        // tile visible to all waves
    if (kt + 1 < HID / 32) stage(buf ^ 1, (kt + 1) * 32);  // overlap with compute
    const v16bf a0 = load_a_frag(&At[buf][(wm)      * LSTR], LSTR);
    const v16bf a1 = load_a_frag(&At[buf][(wm + 16) * LSTR], LSTR);
    const v16bf b0 = load_b_fragT(&Bt[buf][(wn)      * LSTR], LSTR);
    const v16bf b1 = load_b_fragT(&Bt[buf][(wn + 16) * LSTR], LSTR);
    acc[0][0] = WMMA_BF16(a0, b0, acc[0][0]);
    acc[0][1] = WMMA_BF16(a0, b1, acc[0][1]);
    acc[1][0] = WMMA_BF16(a1, b0, acc[1][0]);
    acc[1][1] = WMMA_BF16(a1, b1, acc[1][1]);
  }

  // Epilogue: bias + scatter.  n -> (head, {q,k,v}, d); m -> (b, s).
#pragma unroll
  for (int tm = 0; tm < 2; ++tm) {
#pragma unroll
    for (int tn = 0; tn < 2; ++tn) {
      const int ncol  = n0 + wn + tn * 16 + (lane & 15);
      const float bv  = bias[ncol];
      const int head  = ncol / 192;
      const int rr    = ncol % 192;
      const int which = rr >> 6;            // 0=Q 1=K 2=V
      const int d     = rr & 63;
      const int rowbase = m0 + wm + tm * 16 + ((lane >> 4) << 3);
      const int bb = rowbase >> 10;
      const int s0 = rowbase & (SEQ - 1);
      const int bh = bb * NHEAD + head;
      if (which == 2) {                     // V^T: [bh][d][s], s0 % 8 == 0
        v8bf pv;
#pragma unroll
        for (int v = 0; v < 8; ++v) pv[v] = (bf16)(acc[tm][tn][v] + bv);
        *(v8bf*)(Vt + ((size_t)bh * HD + d) * SEQ + s0) = pv;
      } else {
        bf16* dst = (which == 0 ? Qbf : Kbf) + ((size_t)bh * SEQ + s0) * HD + d;
#pragma unroll
        for (int v = 0; v < 8; ++v) dst[(size_t)v * HD] = (bf16)(acc[tm][tn][v] + bv);
      }
    }
  }
}

// ---------------------------------------------------------------------------
// Kernel 2: flash attention.  Grid (S/128, NH, B), 256 threads = 8 waves,
// each wave owns 16 q-rows; K / V^T tiles async-staged, double-buffered.
// Reference semantics: scores *= sqrt(HD); mask!=0 -> -10000; softmax; P@V.
// ---------------------------------------------------------------------------
__global__ __launch_bounds__(256)
void attn_kernel(const bf16* __restrict__ Qbf, const bf16* __restrict__ Kbf,
                 const bf16* __restrict__ VtG, const int* __restrict__ mask,
                 float* __restrict__ out) {
  constexpr int TSTR = 72;                  // padded tile stride (conflict-free)
  constexpr int PSTR = 72;
  __shared__ bf16 Kt[2][64 * TSTR];         // [krow][d]   (B-layout for QK^T)
  __shared__ bf16 Vs[2][64 * TSTR];         // [d][k]      (B-layout for PV)
  __shared__ bf16 Pb[8][16 * PSTR];         // per-wave probs scratch
  const int t    = threadIdx.x;
  const int lane = t & 31;
  const int wave = t >> 5;
  const int b = blockIdx.z, h = blockIdx.y;
  const int q0 = blockIdx.x * 128 + wave * 16;
  const size_t bh = (size_t)(b * NHEAD + h);
  const bf16* Qp = Qbf + bh * (size_t)(SEQ * HD);
  const bf16* Kp = Kbf + bh * (size_t)(SEQ * HD);
  const bf16* Vp = VtG + bh * (size_t)(SEQ * HD);
  const int*  mrow = mask + b * SEQ;
  const int r  = lane & 15;
  const int c0 = (lane >> 4) * 8;

  // Stage K tile (64 s-rows x 64 d) and V^T tile (64 d-rows x 64 k): 64 B/thread.
  auto stage = [&](int buf, int kbase) {
#pragma unroll
    for (int i = 0; i < 2; ++i) {
      const int c = t + i * 256;            // 0..511
      const int row = c >> 3, cq = (c & 7) * 8;
      async_cp16(&Kt[buf][row * TSTR + cq], Kp + (size_t)(kbase + row) * HD + cq);
    }
#pragma unroll
    for (int i = 0; i < 2; ++i) {
      const int c = t + i * 256;
      const int row = c >> 3, cq = (c & 7) * 8;
      async_cp16(&Vs[buf][row * TSTR + cq], Vp + (size_t)row * SEQ + kbase + cq);
    }
  };

  // Q A-fragments (16 rows x HD=64, two K=32 chunks) in registers.
  v16bf aq[2];
#pragma unroll
  for (int ch = 0; ch < 2; ++ch)
    aq[ch] = load_a_frag(Qp + (size_t)q0 * HD + ch * 32, HD);

  float m_[8], l_[8];
  v8f o[4];
#pragma unroll
  for (int v = 0; v < 8; ++v) { m_[v] = -3.0e38f; l_[v] = 0.f; }
#pragma unroll
  for (int dt = 0; dt < 4; ++dt)
#pragma unroll
    for (int v = 0; v < 8; ++v) o[dt][v] = 0.f;

  bf16* P = &Pb[wave][0];

  stage(0, 0);
  for (int kt = 0; kt < SEQ / 64; ++kt) {
    const int buf   = kt & 1;
    const int kbase = kt * 64;
    wait_async0();
    __syncthreads();
    if (kt + 1 < SEQ / 64) stage(buf ^ 1, kbase + 64);

    // scores: 4 x (16x16) tiles, 2 WMMAs each (HD=64 = two K=32 chunks).
    v8f sc[4];
#pragma unroll
    for (int nt = 0; nt < 4; ++nt) {
      const v16bf bk0 = load_b_fragT(&Kt[buf][(nt * 16) * TSTR],      TSTR);
      const v16bf bk1 = load_b_fragT(&Kt[buf][(nt * 16) * TSTR + 32], TSTR);
      v8f c;
#pragma unroll
      for (int v = 0; v < 8; ++v) c[v] = 0.f;
      c = WMMA_BF16(aq[0], bk0, c);
      c = WMMA_BF16(aq[1], bk1, c);
      const int masked = mrow[kbase + nt * 16 + r];
#pragma unroll
      for (int v = 0; v < 8; ++v) c[v] = masked ? -10000.f : c[v] * 8.0f;
      sc[nt] = c;
    }
    // Online softmax (q-row's 16 cols in one 16-lane half; xor 1/2/4/8 stays in-half).
#pragma unroll
    for (int v = 0; v < 8; ++v) {
      float tmax = fmaxf(fmaxf(sc[0][v], sc[1][v]), fmaxf(sc[2][v], sc[3][v]));
      tmax = fmaxf(tmax, __shfl_xor(tmax, 1));
      tmax = fmaxf(tmax, __shfl_xor(tmax, 2));
      tmax = fmaxf(tmax, __shfl_xor(tmax, 4));
      tmax = fmaxf(tmax, __shfl_xor(tmax, 8));
      const float nm   = fmaxf(m_[v], tmax);
      const float corr = __expf(m_[v] - nm);
      m_[v] = nm;
      float rs = 0.f;
#pragma unroll
      for (int nt = 0; nt < 4; ++nt) {
        const float p = __expf(sc[nt][v] - nm);
        sc[nt][v] = p;
        rs += p;
      }
      rs += __shfl_xor(rs, 1);
      rs += __shfl_xor(rs, 2);
      rs += __shfl_xor(rs, 4);
      rs += __shfl_xor(rs, 8);
      l_[v] = l_[v] * corr + rs;
#pragma unroll
      for (int dt = 0; dt < 4; ++dt) o[dt][v] *= corr;
    }
    // Probs D-layout -> LDS -> A-layout (per-wave buffer; same-wave DS in-order).
    const int prow = (lane >> 4) << 3;
#pragma unroll
    for (int nt = 0; nt < 4; ++nt)
#pragma unroll
      for (int v = 0; v < 8; ++v)
        P[(prow + v) * PSTR + nt * 16 + r] = (bf16)sc[nt][v];
    asm volatile("s_wait_dscnt 0x0" ::: "memory");
    // O(16x64) += P . V  (V^T tile is N-major for B-fragments).
#pragma unroll
    for (int ch = 0; ch < 2; ++ch) {
      const v16bf ap = load_a_frag(P + ch * 32, PSTR);
#pragma unroll
      for (int dt = 0; dt < 4; ++dt) {
        const v16bf bv = load_b_fragT(&Vs[buf][(dt * 16) * TSTR + ch * 32], TSTR);
        o[dt] = WMMA_BF16(ap, bv, o[dt]);
      }
    }
  }
  // Normalize, store fp32 context [b][s][h*HD + d].
#pragma unroll
  for (int dt = 0; dt < 4; ++dt) {
#pragma unroll
    for (int v = 0; v < 8; ++v) {
      const int row = q0 + ((lane >> 4) << 3) + v;
      out[((size_t)(b * SEQ + row)) * HID + h * HD + dt * 16 + r] = o[dt][v] / l_[v];
    }
  }
}

// ---------------------------------------------------------------------------
extern "C" void kernel_launch(void* const* d_in, const int* in_sizes, int n_in,
                              void* d_out, int out_size, void* d_ws, size_t ws_size,
                              hipStream_t stream) {
  const float* hs   = (const float*)d_in[0];   // (8,1024,1024) f32
  const int*   mask = (const int*)  d_in[1];   // (8,1,1,1024) i32
  const float* w    = (const float*)d_in[2];   // (1024,3072) f32
  const float* bias = (const float*)d_in[3];   // (3072,) f32
  float* out = (float*)d_out;                  // (8,1024,1024) f32

  // Workspace: Q, K, V^T, HS(bf16), W^T(bf16)  ->  ~74 MB total.
  const size_t per = (size_t)8 * NHEAD * SEQ * HD;   // 8,388,608 elements
  bf16* Qbf = (bf16*)d_ws;
  bf16* Kbf = Qbf + per;
  bf16* Vt  = Kbf + per;
  bf16* Abf = Vt + per;                        // 8192 x 1024
  bf16* Wt  = Abf + per;                       // 3072 x 1024 (n-major)

  cvt_hs_kernel<<<dim3(4096), 256, 0, stream>>>(hs, Abf);
  wt_kernel<<<dim3(N3 / 64, HID / 64), 256, 0, stream>>>(w, Wt);

  qkv_gemm_kernel<<<dim3(N3 / 64, (8 * SEQ) / 64), 128, 0, stream>>>(
      Abf, Wt, bias, Qbf, Kbf, Vt);

  attn_kernel<<<dim3(SEQ / 128, NHEAD, 8), 256, 0, stream>>>(
      Qbf, Kbf, Vt, mask, out);
}
